// MHSASDPA_81028853006536
// MI455X (gfx1250) — compile-verified, matrix operand order checked by reading, not run on hardware
//
#include <hip/hip_runtime.h>

// ---------------- CDNA5 WMMA types ----------------
typedef __attribute__((ext_vector_type(16))) _Float16 v16h;
typedef __attribute__((ext_vector_type(8)))  _Float16 v8h;
typedef __attribute__((ext_vector_type(8)))  float    v8f;

#define WMMA_F16(A, B, C) \
  __builtin_amdgcn_wmma_f32_16x16x32_f16(false, (A), false, (B), (short)0, (C), false, false)

// Problem constants
#define LW    64          // window length (8x8)
#define DIM   256
#define HEADS 8
#define HD    32          // head dim
#define NB    4096        // batch (windows)

// LDS strides (halfs), padded for bank-conflict-free ds_load_b128
#define SX_STR 264        // 256 + 8
#define QK_STR 40         // 32 + 8
#define VT_STR 72         // 64 + 8
#define P_STR  72
#define SO_STR 264

// ---------------------------------------------------------------------------
// A/B fragment loader per CDNA5 16-bit A-matrix layout:
//   lanes 0-15 : K = kb+0..7 (v8h lo) and kb+16..23 (v8h hi)
//   lanes 16-31: caller passes kb += 8 -> K = kb+8..15 / kb+24..31
// Works for LDS (ds_load_b128) and global (global_load_b128) pointers.
// ---------------------------------------------------------------------------
__device__ __forceinline__ v16h load_frag(const _Float16* p) {
  v8h lo = *(const v8h*)(p);
  v8h hi = *(const v8h*)(p + 16);
  v16h f;
#pragma unroll
  for (int i = 0; i < 8; ++i) { f[i] = lo[i]; f[i + 8] = hi[i]; }
  return f;
}

// ---------------------------------------------------------------------------
// Prep kernel: f32->f16 transposed weights into workspace + gather rel bias.
//   WtQ[n][k] = W_qkv[k][n]   (768 x 256 halfs)
//   WtP[n][k] = W_proj[k][n]  (256 x 256 halfs)
//   RB[h][q][t] = rel_table[rel_index[q][t]][h]  (8 x 64 x 64 f32)
// ---------------------------------------------------------------------------
__global__ void prep_kernel(const float* __restrict__ Wqkv,
                            const float* __restrict__ Wproj,
                            const float* __restrict__ rtab,
                            const int*   __restrict__ ridx,
                            _Float16* __restrict__ WtQ,
                            _Float16* __restrict__ WtP,
                            float* __restrict__ RB) {
  const int i = blockIdx.x * 256 + threadIdx.x;   // up to 768*256
  if (i < 768 * 256) {
    const int n = i >> 8, k = i & 255;
    WtQ[n * 256 + k] = (_Float16)Wqkv[k * 768 + n];
  }
  if (i < 256 * 256) {
    const int n = i >> 8, k = i & 255;
    WtP[n * 256 + k] = (_Float16)Wproj[k * 256 + n];
  }
  if (i < HEADS * LW * LW) {
    const int h  = i >> 12;        // / 4096
    const int qt = i & 4095;       // q*64 + t
    RB[i] = rtab[ridx[qt] * HEADS + h];
  }
}

// ---------------------------------------------------------------------------
// Fused window attention: one block per window, 256 threads = 8 waves.
// ---------------------------------------------------------------------------
__global__ __launch_bounds__(256, 1) void win_attn_kernel(
    const float* __restrict__ x,
    const float* __restrict__ bqkv,
    const float* __restrict__ bproj,
    const _Float16* __restrict__ WtQ,
    const _Float16* __restrict__ WtP,
    const float* __restrict__ RB,
    float* __restrict__ out) {

  // phase1 x-staging region, reused as softmax P (8 heads x 64 x P_STR)
  __shared__ __align__(16) _Float16 sX[HEADS * LW * P_STR];   // 36864 h = 72 KB? (halfs*2 = 73728 B)
  __shared__ __align__(16) _Float16 sQ[HEADS * LW * QK_STR];  // 20480 h
  __shared__ __align__(16) _Float16 sK[HEADS * LW * QK_STR];  // 20480 h
  __shared__ __align__(16) _Float16 sV[HEADS * HD * VT_STR];  // 18432 h (v transposed: [h][d][t])
  __shared__ __align__(16) _Float16 sO[LW * SO_STR];          // 16896 h

  const int b    = blockIdx.x;
  const int tid  = threadIdx.x;
  const int wave = tid >> 5;
  const int lane = tid & 31;
  const int n15  = lane & 15;
  const int hh   = lane >> 4;      // wave half: 0 or 1
  const int kofs = hh * 8;         // lane-half K offset for fragments

  // ---------------- Phase 0: stage x -> LDS f16 ----------------
  const float* xb = x + (size_t)b * (LW * DIM);
  for (int i = tid; i < LW * DIM; i += 256) {
    const int m = i >> 8, c = i & 255;
    sX[m * SX_STR + c] = (_Float16)xb[i];
  }
  __syncthreads();

  // ---------------- Phase 1: QKV GEMM (64x256 @ 256x768) ----------------
  const float scale = 0.17677669529663687f;   // 32^-0.5
  for (int t = 0; t < 6; ++t) {
    const int nt = wave * 6 + t;
    const int n0 = nt * 16;
    v8f acc[4] = {};
    const _Float16* wcol = WtQ + (size_t)(n0 + n15) * 256;
    for (int k0 = 0; k0 < 256; k0 += 32) {
      const v16h bf = load_frag(wcol + k0 + kofs);
#pragma unroll
      for (int mt = 0; mt < 4; ++mt) {
        const v16h af = load_frag(&sX[(mt * 16 + n15) * SX_STR + k0 + kofs]);
        acc[mt] = WMMA_F16(af, bf, acc[mt]);
      }
    }
    const float bq   = bqkv[n0 + n15];
    const int   sec  = n0 >> 8;          // 0=q 1=k 2=v
    const int   head = (n0 & 255) >> 5;
    const int   d0   = n0 & 31;          // 0 or 16
#pragma unroll
    for (int mt = 0; mt < 4; ++mt) {
#pragma unroll
      for (int j = 0; j < 8; ++j) {
        const int m = mt * 16 + j + hh * 8;
        const float vl = acc[mt][j] + bq;
        if (sec == 0)
          sQ[head * (LW * QK_STR) + m * QK_STR + d0 + n15] = (_Float16)(vl * scale);
        else if (sec == 1)
          sK[head * (LW * QK_STR) + m * QK_STR + d0 + n15] = (_Float16)vl;
        else
          sV[head * (HD * VT_STR) + (d0 + n15) * VT_STR + m] = (_Float16)vl;  // v^T
      }
    }
  }
  __syncthreads();

  // ---------------- Phase 2: attention, wave == head ----------------
  const int h = wave;
  _Float16* P = sX + h * (LW * P_STR);                 // reuse x-staging region
  const _Float16* Qh  = sQ + h * (LW * QK_STR);
  const _Float16* Kh  = sK + h * (LW * QK_STR);
  const _Float16* Vh  = sV + h * (HD * VT_STR);
  const float*    RBh = RB + h * (LW * LW);

  for (int mt = 0; mt < 4; ++mt) {
    const v16h aq = load_frag(&Qh[(mt * 16 + n15) * QK_STR + kofs]);
    v8f s[4];
#pragma unroll
    for (int nt = 0; nt < 4; ++nt) {
      const v16h bk = load_frag(&Kh[(nt * 16 + n15) * QK_STR + kofs]);
      v8f z = {};
      s[nt] = WMMA_F16(aq, bk, z);
    }
    // softmax across rows; row m lives in vreg j across lanes of one 16-group
#pragma unroll
    for (int j = 0; j < 8; ++j) {
      const int m = mt * 16 + j + hh * 8;
      float e[4];
#pragma unroll
      for (int nt = 0; nt < 4; ++nt)
        e[nt] = s[nt][j] + RBh[m * LW + nt * 16 + n15];
      float mx = fmaxf(fmaxf(e[0], e[1]), fmaxf(e[2], e[3]));
#pragma unroll
      for (int off = 1; off < 16; off <<= 1)
        mx = fmaxf(mx, __shfl_xor(mx, off, 32));
      float sum = 0.f;
#pragma unroll
      for (int nt = 0; nt < 4; ++nt) { e[nt] = __expf(e[nt] - mx); sum += e[nt]; }
#pragma unroll
      for (int off = 1; off < 16; off <<= 1)
        sum += __shfl_xor(sum, off, 32);
      const float inv = 1.0f / sum;
#pragma unroll
      for (int nt = 0; nt < 4; ++nt)
        P[m * P_STR + nt * 16 + n15] = (_Float16)(e[nt] * inv);
    }
  }

  // P (64x64) @ V (64x32)
#pragma unroll
  for (int mt = 0; mt < 4; ++mt) {
    v8f o[2] = {};
#pragma unroll
    for (int ks = 0; ks < 2; ++ks) {
      const v16h ap = load_frag(&P[(mt * 16 + n15) * P_STR + ks * 32 + kofs]);
#pragma unroll
      for (int nt = 0; nt < 2; ++nt) {
        const v16h bv = load_frag(&Vh[(nt * 16 + n15) * VT_STR + ks * 32 + kofs]);
        o[nt] = WMMA_F16(ap, bv, o[nt]);
      }
    }
#pragma unroll
    for (int nt = 0; nt < 2; ++nt) {
#pragma unroll
      for (int j = 0; j < 8; ++j) {
        const int m = mt * 16 + j + hh * 8;
        sO[m * SO_STR + h * HD + nt * 16 + n15] = (_Float16)o[nt][j];
      }
    }
  }
  __syncthreads();

  // ---------------- Phase 3: output projection (64x256 @ 256x256) ----------------
  float* outb = out + (size_t)b * (LW * DIM);
#pragma unroll
  for (int t = 0; t < 2; ++t) {
    const int n0 = (wave * 2 + t) * 16;
    const _Float16* wcol = WtP + (size_t)(n0 + n15) * 256;
    v8f acc[4] = {};
    for (int k0 = 0; k0 < 256; k0 += 32) {
      const v16h bf = load_frag(wcol + k0 + kofs);
#pragma unroll
      for (int mt = 0; mt < 4; ++mt) {
        const v16h af = load_frag(&sO[(mt * 16 + n15) * SO_STR + k0 + kofs]);
        acc[mt] = WMMA_F16(af, bf, acc[mt]);
      }
    }
    const float bp = bproj[n0 + n15];
#pragma unroll
    for (int mt = 0; mt < 4; ++mt) {
#pragma unroll
      for (int j = 0; j < 8; ++j) {
        const int m = mt * 16 + j + hh * 8;
        outb[m * DIM + n0 + n15] = acc[mt][j] + bp;
      }
    }
  }
}

// ---------------------------------------------------------------------------
extern "C" void kernel_launch(void* const* d_in, const int* in_sizes, int n_in,
                              void* d_out, int out_size, void* d_ws, size_t ws_size,
                              hipStream_t stream) {
  const float* x     = (const float*)d_in[0];
  const float* Wqkv  = (const float*)d_in[1];
  const float* bqkv  = (const float*)d_in[2];
  const float* Wproj = (const float*)d_in[3];
  const float* bproj = (const float*)d_in[4];
  const float* rtab  = (const float*)d_in[5];
  const int*   ridx  = (const int*)d_in[6];

  _Float16* WtQ = (_Float16*)d_ws;                // 768*256 halfs
  _Float16* WtP = WtQ + 768 * 256;                // 256*256 halfs
  float*    RB  = (float*)(WtP + 256 * 256);      // 8*64*64 f32

  prep_kernel<<<768, 256, 0, stream>>>(Wqkv, Wproj, rtab, ridx, WtQ, WtP, RB);
  win_attn_kernel<<<NB, 256, 0, stream>>>(x, bqkv, bproj, WtQ, WtP, RB, (float*)d_out);
}
